// IcoSweepNet_76072460747259
// MI455X (gfx1250) — compile-verified, matrix operand order checked by reading, not run on hardware
//
#include <hip/hip_runtime.h>
#include <math.h>

// Problem constants (from reference)
#define NCAM  4
#define BB    1
#define CC    32
#define VV    40962
#define NDISP 32
#define MM    24576

#define NT    ((VV + 15) / 16)   // 2561 vertex tiles of 16 per cam

typedef __attribute__((ext_vector_type(16))) _Float16 v16h;
typedef __attribute__((ext_vector_type(8)))  float    v8f;

// ---------------------------------------------------------------------------
// Kernel 1: colsum[cam][v] = sum_c feats[cam,0,c,v]  via WMMA.
//   D = ones(16x32) x feat_tile(32x16)  ->  every row of D = column sums.
// One wave handles one 16-vertex tile of one cam. Grid sized exactly so EXEC
// is all-1s at the WMMA (no divergent early-out).
// B operand as loaded here (contiguous k, not AMD's table order): lanes 0-15 hold K=0..15 for N=lane,
// lanes 16-31 hold K=16..31 for N=lane-16; VGPR j packs K=base+2j / base+2j+1.
// ---------------------------------------------------------------------------
__global__ void colsum_wmma_kernel(const float* __restrict__ feats,
                                   float* __restrict__ colsum) {
  const int gwave = blockIdx.x * (blockDim.x >> 5) + (threadIdx.x >> 5);
  const int lane  = threadIdx.x & 31;

  const int cam   = gwave / NT;          // wave-uniform
  const int v0    = (gwave % NT) * 16;   // wave-uniform
  const int n     = lane & 15;           // column (vertex) within tile
  const int kbase = (lane >> 4) << 4;    // 0 for lanes 0-15, 16 for lanes 16-31

  const int v  = v0 + n;
  const int vc = (v < VV) ? v : (VV - 1);   // clamp loads at the ragged edge

  const float* fbase = feats + (size_t)cam * CC * VV + vc;

  v16h b;
#pragma unroll
  for (int i = 0; i < 16; ++i)
    b[i] = (_Float16)fbase[(size_t)(kbase + i) * VV];   // K = kbase+i, N = n

  v16h a;
#pragma unroll
  for (int i = 0; i < 16; ++i)
    a[i] = (_Float16)1.0f;                 // all-ones A: layout-invariant

  v8f c = {};
  // (neg_a, A, neg_b, B, c_mod, C, reuse_a, reuse_b)
  c = __builtin_amdgcn_wmma_f32_16x16x32_f16(false, a, false, b,
                                             (short)0, c, false, false);

  // D VGPR0: lanes 0-15 -> (M=0, N=lane). All rows equal; lanes 0-15 commit.
  if (lane < 16 && v < VV)
    colsum[cam * VV + v] = c[0];
}

// ---------------------------------------------------------------------------
// Kernel 2: zero the [NDISP, V] cost accumulator (ws is poisoned by harness).
// ---------------------------------------------------------------------------
__global__ void zero_kernel(float* __restrict__ p, int n) {
  int i = blockIdx.x * blockDim.x + threadIdx.x;
  if (i < n) p[i] = 0.0f;
}

// ---------------------------------------------------------------------------
// Kernel 3: per (cam,d,m): s = sum_t w[t]*colsum[cam, ids[t]]; scatter with
// sorted-run "last write wins" (matches .set duplicate semantics), summed
// across cams via f32 global atomics.
// ---------------------------------------------------------------------------
__global__ void gather_scatter_kernel(const float* __restrict__ weights,
                                      const int*   __restrict__ vertex_ids,
                                      const int*   __restrict__ inside_ids,
                                      const float* __restrict__ colsum,
                                      float*       __restrict__ cost) {
  const long idx = (long)blockIdx.x * blockDim.x + threadIdx.x;
  const long TOT = (long)NCAM * NDISP * MM;
  if (idx >= TOT) return;

  const int  m   = (int)(idx % MM);
  const long cd  = idx / MM;            // cam*NDISP + d
  const int  d   = (int)(cd % NDISP);
  const int  cam = (int)(cd / NDISP);

  const long wb = cd * (3L * MM) + m;   // [cam][d][t][m] with t stride M
  const float w0 = weights[wb];
  const float w1 = weights[wb + MM];
  const float w2 = weights[wb + 2 * MM];
  const int  i0 = vertex_ids[wb];
  const int  i1 = vertex_ids[wb + MM];
  const int  i2 = vertex_ids[wb + 2 * MM];

  const float* cs = colsum + (long)cam * VV;   // 160KB/cam table, L2-resident
  const float s = w0 * cs[i0] + w1 * cs[i1] + w2 * cs[i2];

  const long ib  = cd * (long)MM + m;
  const int  vtx = inside_ids[ib];
  // inside_ids sorted along m: commit only the last element of an equal-run.
  const bool last = (m == MM - 1) || (inside_ids[ib + 1] != vtx);
  if (last)
    atomicAdd(&cost[(long)d * VV + vtx], s);
}

// ---------------------------------------------------------------------------
// Kernel 4: per-vertex channel-mean scale, softmax over NDISP, soft-argmax.
// ---------------------------------------------------------------------------
__global__ void softargmax_kernel(const float* __restrict__ cost,
                                  float* __restrict__ out) {
  const int v = blockIdx.x * blockDim.x + threadIdx.x;
  if (v >= VV) return;

  const float inv_ch = 1.0f / (float)(NCAM * CC);
  float x[NDISP];
  float mx = -INFINITY;
#pragma unroll
  for (int d = 0; d < NDISP; ++d) {
    x[d] = cost[(long)d * VV + v] * inv_ch;
    mx = fmaxf(mx, x[d]);
  }
  float se = 0.0f, sd = 0.0f;
#pragma unroll
  for (int d = 0; d < NDISP; ++d) {
    const float e = __expf(x[d] - mx);
    se += e;
    sd += e * (float)d;
  }
  out[v] = sd / se;
}

// ---------------------------------------------------------------------------
// Launch. d_in: [0]=feats f32, [1]=weights f32, [2]=vertex_ids i32,
// [3]=inside_ids i32. d_out: B*V = 40962 f32.
// Workspace: colsum (NCAM*V f32) | cost (NDISP*V f32)  ~ 5.9 MB.
// ---------------------------------------------------------------------------
extern "C" void kernel_launch(void* const* d_in, const int* in_sizes, int n_in,
                              void* d_out, int out_size, void* d_ws, size_t ws_size,
                              hipStream_t stream) {
  const float* feats      = (const float*)d_in[0];
  const float* weights    = (const float*)d_in[1];
  const int*   vertex_ids = (const int*)d_in[2];
  const int*   inside_ids = (const int*)d_in[3];
  float*       out        = (float*)d_out;

  float* colsum = (float*)d_ws;                 // NCAM*V floats
  float* cost   = colsum + (size_t)NCAM * VV;   // NDISP*V floats

  // 1) zero the cost accumulator
  {
    const int n = NDISP * VV;
    zero_kernel<<<(n + 255) / 256, 256, 0, stream>>>(cost, n);
  }
  // 2) channel reduction via WMMA: exactly NCAM*NT waves, 4 waves/block
  {
    const int waves  = NCAM * NT;      // 10244 (divisible by 4)
    const int blocks = waves / 4;      // 2561 blocks of 128 threads
    colsum_wmma_kernel<<<blocks, 128, 0, stream>>>(feats, colsum);
  }
  // 3) gather + sorted-run scatter
  {
    const long tot = (long)NCAM * NDISP * MM;   // 3,145,728
    gather_scatter_kernel<<<(int)((tot + 255) / 256), 256, 0, stream>>>(
        weights, vertex_ids, inside_ids, colsum, cost);
  }
  // 4) soft-argmax over disparity
  {
    softargmax_kernel<<<(VV + 255) / 256, 256, 0, stream>>>(cost, out);
  }
}
